// SentenceLSTM_57612691308982
// MI455X (gfx1250) — matrix operand order repaired and checked by reading, not compiled, not run on hardware
//
#include <hip/hip_runtime.h>
#include <hip/hip_bf16.h>

typedef __bf16 bf16;
typedef __attribute__((ext_vector_type(16))) __bf16 bf16x16;
typedef __attribute__((ext_vector_type(8)))  __bf16 bf16x8;
typedef __attribute__((ext_vector_type(8)))  float  f32x8;

#define B_ 256
#define N_ 196
#define D_ 1024
#define HID_ 1024
#define ATT_ 256
#define SIN_ 1024
#define WID_ 1024
#define ISD_ 256
#define S_ 10

// ---------------------------------------------------------------------------
// WMMA bf16 GEMM:  C[M,N] = act( A[M,K](bf16,row-major) @ B[N,K]^T (bf16) + bias[N] )
// One wave -> 32x32 tile (2x2 of 16x16 WMMA). Block (256 thr, 8 waves) -> 64x128.
// grid = (M/64, N/128).
// A-frag layout (16-bit A 16x32): lane m=lane&15 row, half=lane>>4,
//   elems 0..7  = K(k+8*half .. +7), elems 8..15 = K(k+16+8*half .. +7)
// B-frag layout (16-bit B 32x16): lane n=lane&15 col, same K order as A-frag
// C/D: VGPR v -> row = 8*half + v, col = lane&15
// ---------------------------------------------------------------------------
__device__ __forceinline__ bf16x16 load_a_frag(const bf16* __restrict__ rowp, int k, int half) {
  const bf16x8 lo = *(const bf16x8*)(rowp + k + 8 * half);
  const bf16x8 hi = *(const bf16x8*)(rowp + k + 16 + 8 * half);
  bf16x16 r;
#pragma unroll
  for (int i = 0; i < 8; ++i) { r[i] = lo[i]; r[i + 8] = hi[i]; }
  return r;
}

template <typename OutT, bool TANH>
__device__ __forceinline__ void store_frag(const f32x8& c, OutT* __restrict__ C, int ldc,
                                           int row, int col, const float* __restrict__ bias) {
  const float bv = bias ? bias[col] : 0.0f;
#pragma unroll
  for (int v = 0; v < 8; ++v) {
    float x = c[v] + bv;
    if (TANH) x = tanhf(x);
    C[(size_t)(row + v) * ldc + col] = (OutT)x;
  }
}

template <typename OutT, bool TANH>
__global__ __launch_bounds__(256) void gemm_bf16_kernel(
    const bf16* __restrict__ A, int lda, const bf16* __restrict__ Bm, int ldb,
    const float* __restrict__ bias, OutT* __restrict__ C, int ldc, int K) {
  const int lane = threadIdx.x & 31;
  const int wave = threadIdx.x >> 5;
  const int wm = wave >> 2;            // 0..1
  const int wn = wave & 3;             // 0..3
  const int row0 = blockIdx.x * 64 + wm * 32;
  const int col0 = blockIdx.y * 128 + wn * 32;
  const int half = lane >> 4;
  const int l15 = lane & 15;

  const bf16* pA0 = A + (size_t)(row0 + l15) * lda;
  const bf16* pA1 = pA0 + (size_t)16 * lda;
  const bf16* pB0 = Bm + (size_t)(col0 + l15) * ldb;
  const bf16* pB1 = pB0 + (size_t)16 * ldb;

  f32x8 c00 = {0.f,0.f,0.f,0.f,0.f,0.f,0.f,0.f};
  f32x8 c01 = c00, c10 = c00, c11 = c00;

  for (int k = 0; k < K; k += 32) {
    const bf16x16 a0 = load_a_frag(pA0, k, half);
    const bf16x16 a1 = load_a_frag(pA1, k, half);
    const bf16x16 b0 = load_a_frag(pB0, k, half);
    const bf16x16 b1 = load_a_frag(pB1, k, half);
    c00 = __builtin_amdgcn_wmma_f32_16x16x32_bf16(false, a0, false, b0, (short)0, c00, false, false);
    c01 = __builtin_amdgcn_wmma_f32_16x16x32_bf16(false, a0, false, b1, (short)0, c01, false, false);
    c10 = __builtin_amdgcn_wmma_f32_16x16x32_bf16(false, a1, false, b0, (short)0, c10, false, false);
    c11 = __builtin_amdgcn_wmma_f32_16x16x32_bf16(false, a1, false, b1, (short)0, c11, false, false);
  }

  const int r0 = row0 + half * 8;
  const int n0 = col0 + l15;
  store_frag<OutT, TANH>(c00, C, ldc, r0,      n0,      bias);
  store_frag<OutT, TANH>(c01, C, ldc, r0,      n0 + 16, bias);
  store_frag<OutT, TANH>(c10, C, ldc, r0 + 16, n0,      bias);
  store_frag<OutT, TANH>(c11, C, ldc, r0 + 16, n0 + 16, bias);
}

// ---------------------------------------------------------------------------
// Setup / pointwise kernels
// ---------------------------------------------------------------------------
__global__ void cvt_bf16_kernel(const float* __restrict__ src, bf16* __restrict__ dst, int n) {
  int i = blockIdx.x * 256 + threadIdx.x;
  if (i < n) dst[i] = (bf16)src[i];
}

// dst[n, 0:K0]=a[n,:], dst[n, K0:2K0]=b[n,:]   (grid: (K0/256, Nrows))
__global__ void concat2_bf16_kernel(const float* __restrict__ a, const float* __restrict__ b,
                                    bf16* __restrict__ dst, int K0) {
  int n = blockIdx.y;
  int k = blockIdx.x * 256 + threadIdx.x;
  dst[(size_t)n * 2 * K0 + k]      = (bf16)a[(size_t)n * K0 + k];
  dst[(size_t)n * 2 * K0 + K0 + k] = (bf16)b[(size_t)n * K0 + k];
}

__global__ void addvec_kernel(const float* __restrict__ a, const float* __restrict__ b,
                              float* __restrict__ d, int n) {
  int i = blockIdx.x * 256 + threadIdx.x;
  if (i < n) d[i] = a[i] + b[i];
}

// zero c [B*1024] f32 and X [B*4096] bf16 ; grid 1024 x 256 thr
__global__ void zero_state_kernel(float* __restrict__ c, bf16* __restrict__ X) {
  int i = blockIdx.x * 256 + threadIdx.x;   // 0 .. 262143
  c[i] = 0.0f;
#pragma unroll
  for (int r = 0; r < 4; ++r) X[(size_t)i + (size_t)r * 262144] = (bf16)0.0f;
}

// logits[b,n] = fb + sum_a tanh(visatt[b,n,a] + dec[b,a]) * fw[a]   (one wave per (b,n))
__global__ __launch_bounds__(256) void logits_kernel(
    const float* __restrict__ visatt, const float* __restrict__ dec,
    const float* __restrict__ fw, const float* __restrict__ fb, float* __restrict__ logits) {
  const int gw = (blockIdx.x * 256 + threadIdx.x) >> 5;
  const int lane = threadIdx.x & 31;
  const int b = gw / N_;
  const float* va = visatt + (size_t)gw * ATT_;
  const float* dd = dec + (size_t)b * ATT_;
  float s = 0.0f;
#pragma unroll
  for (int a = lane; a < ATT_; a += 32) s += tanhf(va[a] + dd[a]) * fw[a];
#pragma unroll
  for (int off = 16; off; off >>= 1) s += __shfl_xor(s, off, 32);
  if (lane == 0) logits[gw] = s + fb[0];
}

// in-place softmax over N_=196 per batch row; block per b, 256 threads
__global__ __launch_bounds__(256) void softmax_kernel(float* __restrict__ sl) {
  __shared__ float red[8];
  __shared__ float mval, sval;
  const int b = blockIdx.x, t = threadIdx.x;
  float* row = sl + (size_t)b * N_;
  const float v = (t < N_) ? row[t] : -3.4e38f;
  float m = v;
#pragma unroll
  for (int off = 16; off; off >>= 1) m = fmaxf(m, __shfl_xor(m, off, 32));
  if ((t & 31) == 0) red[t >> 5] = m;
  __syncthreads();
  if (t == 0) { float mm = red[0]; for (int i = 1; i < 8; ++i) mm = fmaxf(mm, red[i]); mval = mm; }
  __syncthreads();
  const float e = (t < N_) ? __expf(v - mval) : 0.0f;
  float s = e;
#pragma unroll
  for (int off = 16; off; off >>= 1) s += __shfl_xor(s, off, 32);
  if ((t & 31) == 0) red[t >> 5] = s;
  __syncthreads();
  if (t == 0) { float ss = 0.f; for (int i = 0; i < 8; ++i) ss += red[i]; sval = ss; }
  __syncthreads();
  if (t < N_) row[t] = e / sval;
}

// attout_bf[b,d] = sum_n scores[b,n] * vis_bf[b,n,d]  ; grid (D/256, B)
__global__ __launch_bounds__(256) void attout_kernel(
    const float* __restrict__ scores, const bf16* __restrict__ vis_bf, bf16* __restrict__ attout) {
  const int b = blockIdx.y;
  const int d = blockIdx.x * 256 + threadIdx.x;
  const bf16* vb = vis_bf + ((size_t)b * N_) * D_ + d;
  const float* sc = scores + (size_t)b * N_;
  float acc = 0.0f;
  for (int n = 0; n < N_; ++n) acc += sc[n] * (float)vb[(size_t)n * D_];
  attout[(size_t)b * D_ + d] = (bf16)acc;
}

__device__ __forceinline__ float sigmf(float x) { return 1.0f / (1.0f + __expf(-x)); }

// LSTM pointwise; writes h_new (bf16) to X col0 and col3 ; grid (4, B)
__global__ __launch_bounds__(256) void lstm_kernel(
    const float* __restrict__ gates, float* __restrict__ c, bf16* __restrict__ X) {
  const int b = blockIdx.y;
  const int j = blockIdx.x * 256 + threadIdx.x;
  const float* g = gates + (size_t)b * 4 * HID_;
  const float gi = g[j], gf = g[j + HID_], gg = g[j + 2 * HID_], go = g[j + 3 * HID_];
  const float cn = sigmf(gf) * c[(size_t)b * HID_ + j] + sigmf(gi) * tanhf(gg);
  const float h = sigmf(go) * tanhf(cn);
  c[(size_t)b * HID_ + j] = cn;
  const bf16 hb = (bf16)h;
  X[(size_t)b * 4096 + j] = hb;          // col0: h_new
  X[(size_t)b * 4096 + 3072 + j] = hb;   // col3: h_new copy (for stop cat)
}

// h_prev <- h_new ; grid (4, B)
__global__ __launch_bounds__(256) void promote_kernel(bf16* __restrict__ X) {
  const int b = blockIdx.y;
  const int j = blockIdx.x * 256 + threadIdx.x;
  X[(size_t)b * 4096 + 2048 + j] = X[(size_t)b * 4096 + j];
}

// ps[b, t, :] = ppre[b,:] @ final_w^T + final_b ; one thread per b
__global__ __launch_bounds__(256) void stopfinal_kernel(
    const float* __restrict__ ppre, const float* __restrict__ fw, const float* __restrict__ fb,
    float* __restrict__ ps, int t) {
  const int b = threadIdx.x;
  const float* p = ppre + (size_t)b * ISD_;
  float s0 = fb[0], s1 = fb[1];
  for (int k = 0; k < ISD_; ++k) {
    const float v = p[k];
    s0 += v * fw[k];
    s1 += v * fw[ISD_ + k];
  }
  ps[(size_t)b * S_ * 2 + t * 2 + 0] = s0;
  ps[(size_t)b * S_ * 2 + t * 2 + 1] = s1;
}

// ---------------------------------------------------------------------------
extern "C" void kernel_launch(void* const* d_in, const int* in_sizes, int n_in,
                              void* d_out, int out_size, void* d_ws, size_t ws_size,
                              hipStream_t stream) {
  const float* vis_enc      = (const float*)d_in[0];
  // d_in[1] = sentences (int64) -- unused by the reference math
  const float* enc_att_w    = (const float*)d_in[2];
  const float* enc_att_b    = (const float*)d_in[3];
  const float* dec_att_w    = (const float*)d_in[4];
  const float* dec_att_b    = (const float*)d_in[5];
  const float* full_att_w   = (const float*)d_in[6];
  const float* full_att_b   = (const float*)d_in[7];
  const float* ctx_w        = (const float*)d_in[8];
  const float* ctx_b        = (const float*)d_in[9];
  const float* W_ih         = (const float*)d_in[10];
  const float* b_ih         = (const float*)d_in[11];
  const float* W_hh         = (const float*)d_in[12];
  const float* b_hh         = (const float*)d_in[13];
  const float* topic_hid_w  = (const float*)d_in[14];
  const float* topic_hid_b  = (const float*)d_in[15];
  const float* topic_ctx_w  = (const float*)d_in[16];
  const float* topic_ctx_b  = (const float*)d_in[17];
  const float* stop_prev_w  = (const float*)d_in[18];
  const float* stop_prev_b  = (const float*)d_in[19];
  const float* stop_cur_w   = (const float*)d_in[20];
  const float* stop_cur_b   = (const float*)d_in[21];
  const float* final_stop_w = (const float*)d_in[22];
  const float* final_stop_b = (const float*)d_in[23];

  float* topics = (float*)d_out;                            // [B,S,WID]
  float* ps     = topics + (size_t)B_ * S_ * WID_;          // [B,S,2]

  // ---- workspace carve ----
  char* w = (char*)d_ws;
  auto carve = [&](size_t bytes) -> char* {
    char* p = w; w += (bytes + 255) & ~(size_t)255; return p;
  };
  bf16*  vis_bf  = (bf16*)carve((size_t)B_ * N_ * D_ * 2);        // 102.8 MB
  float* visatt  = (float*)carve((size_t)B_ * N_ * ATT_ * 4);     // 51.4 MB
  bf16*  encW    = (bf16*)carve((size_t)ATT_ * D_ * 2);
  bf16*  decW    = (bf16*)carve((size_t)ATT_ * HID_ * 2);
  bf16*  ctxW    = (bf16*)carve((size_t)SIN_ * D_ * 2);
  bf16*  gatesW  = (bf16*)carve((size_t)4 * HID_ * 2048 * 2);     // [W_ih|W_hh]
  bf16*  topicW  = (bf16*)carve((size_t)WID_ * 2048 * 2);         // [hid|ctx]
  bf16*  stopW   = (bf16*)carve((size_t)ISD_ * 2048 * 2);         // [prev|cur]
  float* gbias   = (float*)carve((size_t)4 * HID_ * 4);
  float* tbias   = (float*)carve((size_t)WID_ * 4);
  float* sbias   = (float*)carve((size_t)ISD_ * 4);
  bf16*  X       = (bf16*)carve((size_t)B_ * 4096 * 2);  // [h_new|ctx|h_prev|h_new2]
  float* cbuf    = (float*)carve((size_t)B_ * HID_ * 4);
  float* gates   = (float*)carve((size_t)B_ * 4 * HID_ * 4);
  float* decbuf  = (float*)carve((size_t)B_ * ATT_ * 4);
  float* scores  = (float*)carve((size_t)B_ * N_ * 4);
  bf16*  attout  = (bf16*)carve((size_t)B_ * D_ * 2);
  float* ppre    = (float*)carve((size_t)B_ * ISD_ * 4);

  // ---- one-time (per launch) setup ----
  {
    int n;
    n = B_ * N_ * D_;  cvt_bf16_kernel<<<(n + 255) / 256, 256, 0, stream>>>(vis_enc, vis_bf, n);
    n = ATT_ * D_;     cvt_bf16_kernel<<<(n + 255) / 256, 256, 0, stream>>>(enc_att_w, encW, n);
    n = ATT_ * HID_;   cvt_bf16_kernel<<<(n + 255) / 256, 256, 0, stream>>>(dec_att_w, decW, n);
    n = SIN_ * D_;     cvt_bf16_kernel<<<(n + 255) / 256, 256, 0, stream>>>(ctx_w, ctxW, n);
    concat2_bf16_kernel<<<dim3(4, 4 * HID_), 256, 0, stream>>>(W_ih, W_hh, gatesW, 1024);
    concat2_bf16_kernel<<<dim3(4, WID_), 256, 0, stream>>>(topic_hid_w, topic_ctx_w, topicW, 1024);
    concat2_bf16_kernel<<<dim3(4, ISD_), 256, 0, stream>>>(stop_prev_w, stop_cur_w, stopW, 1024);
    addvec_kernel<<<(4 * HID_) / 256, 256, 0, stream>>>(b_ih, b_hh, gbias, 4 * HID_);
    addvec_kernel<<<WID_ / 256, 256, 0, stream>>>(topic_hid_b, topic_ctx_b, tbias, WID_);
    addvec_kernel<<<1, 256, 0, stream>>>(stop_prev_b, stop_cur_b, sbias, ISD_);
    zero_state_kernel<<<1024, 256, 0, stream>>>(cbuf, X);
    // visatt = vis_bf[B*N, D] @ encW^T + enc_att_b   (M=50176, N=256, K=1024)
    gemm_bf16_kernel<float, false><<<dim3(784, 2), 256, 0, stream>>>(
        vis_bf, D_, encW, D_, enc_att_b, visatt, ATT_, D_);
  }

  // ---- recurrence ----
  for (int t = 0; t < S_; ++t) {
    // dec = h_prev @ decW^T + b   (M=256,N=256,K=1024), A = X col2
    gemm_bf16_kernel<float, false><<<dim3(4, 2), 256, 0, stream>>>(
        X + 2048, 4096, decW, HID_, dec_att_b, decbuf, ATT_, HID_);
    // logits (one wave per (b,n)) -> scores buffer
    logits_kernel<<<(B_ * N_) / 8, 256, 0, stream>>>(visatt, decbuf, full_att_w, full_att_b, scores);
    softmax_kernel<<<B_, 256, 0, stream>>>(scores);
    attout_kernel<<<dim3(D_ / 256, B_), 256, 0, stream>>>(scores, vis_bf, attout);
    // ctx = attout @ ctxW^T + b  -> bf16 into X col1   (M=256,N=1024,K=1024)
    gemm_bf16_kernel<bf16, false><<<dim3(4, 8), 256, 0, stream>>>(
        attout, D_, ctxW, D_, ctx_b, X + 1024, 4096, D_);
    // gates = [ctx|h_prev] @ [W_ih|W_hh]^T + (b_ih+b_hh)   (M=256,N=4096,K=2048)
    gemm_bf16_kernel<float, false><<<dim3(4, 32), 256, 0, stream>>>(
        X + 1024, 4096, gatesW, 2048, gbias, gates, 4 * HID_, 2048);
    lstm_kernel<<<dim3(4, B_), 256, 0, stream>>>(gates, cbuf, X);
    // topic = tanh([h_new|ctx] @ [hidW|ctxW]^T + b)  -> directly into topics[:, t, :]
    gemm_bf16_kernel<float, true><<<dim3(4, 8), 256, 0, stream>>>(
        X, 4096, topicW, 2048, tbias, topics + (size_t)t * WID_, S_ * WID_, 2048);
    // ppre = tanh([h_prev|h_new] @ [prevW|curW]^T + b)   (M=256,N=256,K=2048)
    gemm_bf16_kernel<float, true><<<dim3(4, 2), 256, 0, stream>>>(
        X + 2048, 4096, stopW, 2048, sbias, ppre, ISD_, 2048);
    stopfinal_kernel<<<1, 256, 0, stream>>>(ppre, final_stop_w, final_stop_b, ps, t);
    promote_kernel<<<dim3(4, B_), 256, 0, stream>>>(X);
  }
}
